// EncoderBlock_51402168598765
// MI455X (gfx1250) — matrix operand order repaired and checked by reading, not compiled, not run on hardware
//
#include <hip/hip_runtime.h>
#include <hip/hip_bf16.h>
#include <cstdint>
#include <cstddef>

typedef _Float16 half_t;
typedef __attribute__((ext_vector_type(2)))  _Float16 half2_t;
typedef __attribute__((ext_vector_type(16))) _Float16 v16h;
typedef __attribute__((ext_vector_type(8)))  float    v8f;

// ---------------------------------------------------------------------------
// CDNA5 async global->LDS copy (ASYNCcnt path), with safe fallback
// builtin signature: (int4 AS(1)*, int4 AS(3)*, Imm offset, Imm cpol)
// ---------------------------------------------------------------------------
#if __has_builtin(__builtin_amdgcn_global_load_async_to_lds_b128)
#define HAVE_ASYNC_LDS 1
#else
#define HAVE_ASYNC_LDS 0
#endif

typedef int v4i_t __attribute__((__vector_size__(16)));
typedef __attribute__((address_space(1))) v4i_t* g4p_t;
typedef __attribute__((address_space(3))) v4i_t* l4p_t;

__device__ __forceinline__ void async_copy_b128(const half_t* gsrc, half_t* ldst) {
#if HAVE_ASYNC_LDS
  __builtin_amdgcn_global_load_async_to_lds_b128(
      (g4p_t)(uintptr_t)gsrc,
      (l4p_t)(uint32_t)(uintptr_t)ldst,
      0, 0);
#else
  *(uint4*)ldst = *(const uint4*)gsrc;
#endif
}

__device__ __forceinline__ void async_wait_all() {
#if HAVE_ASYNC_LDS
#if __has_builtin(__builtin_amdgcn_s_wait_asynccnt)
  __builtin_amdgcn_s_wait_asynccnt(0);
#else
  asm volatile("s_wait_asynccnt 0" ::: "memory");
#endif
#endif
}

// ---------------------------------------------------------------------------
// fp32 -> fp16 elementwise conversion (for weights)
// ---------------------------------------------------------------------------
__global__ __launch_bounds__(256) void cvt_f32_f16_kernel(
    const float* __restrict__ in, half_t* __restrict__ out, int n) {
  int i = blockIdx.x * 256 + threadIdx.x;
  if (i < n) out[i] = (half_t)in[i];
}

// ---------------------------------------------------------------------------
// LayerNorm over D=512 (torch-style: unbiased std, (x-mean)/(std+eps))
// one 256-thread block per row, 2 elements/thread; writes f16
// ---------------------------------------------------------------------------
__global__ __launch_bounds__(256) void layernorm_f16_kernel(
    const float* __restrict__ x, const float* __restrict__ alpha,
    const float* __restrict__ beta, half_t* __restrict__ out) {
  __shared__ float red[8];
  const int row = blockIdx.x;
  const int tid = threadIdx.x;
  const float* xr = x + (size_t)row * 512;
  float v0 = xr[tid];
  float v1 = xr[tid + 256];

  float s = v0 + v1;
  #pragma unroll
  for (int off = 16; off; off >>= 1) s += __shfl_xor(s, off, 32);
  if ((tid & 31) == 0) red[tid >> 5] = s;
  __syncthreads();
  float tot = 0.f;
  #pragma unroll
  for (int i = 0; i < 8; ++i) tot += red[i];
  const float mean = tot * (1.0f / 512.0f);

  float d0 = v0 - mean, d1 = v1 - mean;
  float q2 = d0 * d0 + d1 * d1;
  __syncthreads();
  #pragma unroll
  for (int off = 16; off; off >>= 1) q2 += __shfl_xor(q2, off, 32);
  if ((tid & 31) == 0) red[tid >> 5] = q2;
  __syncthreads();
  float tot2 = 0.f;
  #pragma unroll
  for (int i = 0; i < 8; ++i) tot2 += red[i];
  const float var = tot2 * (1.0f / 511.0f);     // Bessel-corrected
  const float inv = alpha[0] / (sqrtf(var) + 1e-6f);
  const float b0 = beta[0];

  half_t* orow = out + (size_t)row * 512;
  orow[tid]       = (half_t)(d0 * inv + b0);
  orow[tid + 256] = (half_t)(d1 * inv + b0);
}

// ---------------------------------------------------------------------------
// WMMA GEMM:  out[M,N] = A[M,K](f16) * W[N,K]^T(f16) + bias[N]
//             (+ residual[M,N] fp32) (+ReLU); writes f32 and/or f16
// block tile 128x128x64, 8 waves, each wave = 32x64 (2x4 wmma tiles)
// ---------------------------------------------------------------------------
#define GBM 128
#define GBN 128
#define GBK 64

__global__ __launch_bounds__(256) void gemm_wmma_kernel(
    const half_t* __restrict__ A, const half_t* __restrict__ W,
    const float* __restrict__ bias, const float* __restrict__ residual,
    float* __restrict__ outF, half_t* __restrict__ outH,
    int M, int N, int K, int relu) {
  __shared__ alignas(16) half_t lA[GBM * GBK];
  __shared__ alignas(16) half_t lW[GBN * GBK];

  const int tid  = threadIdx.x;
  const int lane = tid & 31;
  const int wave = tid >> 5;
  const int hi   = (lane >> 4) & 1;
  const int ln16 = lane & 15;

  const int bm = blockIdx.y * GBM;
  const int bn = blockIdx.x * GBN;
  const int wm = (wave & 3) * 32;   // 4 waves along M
  const int wn = (wave >> 2) * 64;  // 2 waves along N

  v8f zero{};
  v8f c[2][4];
  #pragma unroll
  for (int i = 0; i < 2; ++i)
    #pragma unroll
    for (int j = 0; j < 4; ++j) c[i][j] = zero;

  for (int k0 = 0; k0 < K; k0 += GBK) {
    __syncthreads();
    // cooperative tile stage: 1024 chunks of 16B each for A and W,
    // issued as async global->LDS DMA (no VGPR round trip)
    #pragma unroll
    for (int t = 0; t < 4; ++t) {
      int chunk = tid + t * 256;
      int r  = chunk >> 3;            // 0..127
      int cc = (chunk & 7) << 3;      // 0..56 halves
      async_copy_b128(&A[(size_t)(bm + r) * K + k0 + cc], &lA[r * GBK + cc]);
      async_copy_b128(&W[(size_t)(bn + r) * K + k0 + cc], &lW[r * GBK + cc]);
    }
    if (k0 + GBK < K) {   // prefetch next K-tile -> global_prefetch_b8
      __builtin_prefetch(&A[(size_t)(bm + (tid >> 1)) * K + k0 + GBK], 0, 1);
      __builtin_prefetch(&W[(size_t)(bn + (tid >> 1)) * K + k0 + GBK], 0, 1);
    }
    async_wait_all();
    __syncthreads();

    #pragma unroll
    for (int ks = 0; ks < GBK; ks += 32) {
      v16h a[2], b[4];
      #pragma unroll
      for (int i = 0; i < 2; ++i) {
        const half_t* p = &lA[(wm + i * 16 + ln16) * GBK + ks];
        #pragma unroll
        for (int v = 0; v < 8; ++v) {
          int kk = (hi << 4) + (v << 1);
          half2_t h2 = *(const half2_t*)&p[kk];
          a[i][2 * v] = h2[0]; a[i][2 * v + 1] = h2[1];
        }
      }
      #pragma unroll
      for (int j = 0; j < 4; ++j) {
        const half_t* p = &lW[(wn + j * 16 + ln16) * GBK + ks + (hi << 4)];
        #pragma unroll
        for (int v = 0; v < 8; ++v) {
          half2_t h2 = *(const half2_t*)&p[2 * v];
          b[j][2 * v] = h2[0]; b[j][2 * v + 1] = h2[1];
        }
      }
      #pragma unroll
      for (int i = 0; i < 2; ++i)
        #pragma unroll
        for (int j = 0; j < 4; ++j)
          c[i][j] = __builtin_amdgcn_wmma_f32_16x16x32_f16(
              false, a[i], false, b[j], (short)0, c[i][j], false, false);
    }
  }

  // epilogue
  #pragma unroll
  for (int i = 0; i < 2; ++i) {
    #pragma unroll
    for (int j = 0; j < 4; ++j) {
      int n = bn + wn + j * 16 + ln16;
      float bv = bias ? bias[n] : 0.0f;
      #pragma unroll
      for (int r = 0; r < 8; ++r) {
        int m = bm + wm + i * 16 + r + (hi << 3);
        float val = c[i][j][r] + bv;
        if (residual) val += residual[(size_t)m * N + n];
        if (relu) val = val > 0.f ? val : 0.f;
        if (outF) outF[(size_t)m * N + n] = val;
        if (outH) outH[(size_t)m * N + n] = (half_t)val;
      }
    }
  }
}

// ---------------------------------------------------------------------------
// Flash-style attention with WMMA.
// q,k,v,ctx layout: [B*S, H*64] f16.  block = (query block of 128, b*h)
// each of 8 waves owns a 16-query tile; K/V tiles (32 keys) staged in LDS.
// ---------------------------------------------------------------------------
#define KT 32

__global__ __launch_bounds__(256) void attn_wmma_kernel(
    const half_t* __restrict__ q, const half_t* __restrict__ k,
    const half_t* __restrict__ v, const int* __restrict__ mask,
    half_t* __restrict__ ctx, int S, int Dm) {
  __shared__ alignas(16) half_t Kt[KT * 64];      // [key][dk]
  __shared__ alignas(16) half_t Vt[64 * KT];      // [dk][key] (transposed)
  __shared__ alignas(16) half_t Pb[8 * 16 * KT];  // per-wave P staging

  const int bh = blockIdx.y;
  const int b  = bh >> 3;
  const int h  = bh & 7;
  const int qbase = blockIdx.x * 128;
  const int tid  = threadIdx.x;
  const int lane = tid & 31;
  const int wave = tid >> 5;
  const int hi   = (lane >> 4) & 1;
  const int ln16 = lane & 15;

  const size_t rowBase = (size_t)b * S;
  const int colBase = h * 64;

  // load this wave's Q fragments (16 queries x 64 dk, 2 k-steps) from global
  v16h aq[2];
  {
    const half_t* qp = q + (rowBase + qbase + wave * 16 + ln16) * Dm + colBase;
    #pragma unroll
    for (int i = 0; i < 2; ++i)
      #pragma unroll
      for (int vv = 0; vv < 8; ++vv) {
        int kk = 32 * i + (hi << 4) + (vv << 1);
        half2_t h2 = *(const half2_t*)&qp[kk];
        aq[i][2 * vv] = h2[0]; aq[i][2 * vv + 1] = h2[1];
      }
  }

  v8f zero{};
  v8f o[4];
  #pragma unroll
  for (int j = 0; j < 4; ++j) o[j] = zero;
  float mrun[8], lrun[8];
  #pragma unroll
  for (int r = 0; r < 8; ++r) { mrun[r] = -1e30f; lrun[r] = 0.f; }

  const float scale = 0.125f;  // 1/sqrt(64)
  const int nkt = S / KT;
  for (int kt = 0; kt < nkt; ++kt) {
    const int key0 = kt * KT;
    __syncthreads();
    {
      // K tile: async DMA 16B/thread; V tile transposed via VGPR scatter
      int r  = tid >> 3;             // key 0..31
      int cc = (tid & 7) << 3;       // dk 0..56
      async_copy_b128(&k[(rowBase + key0 + r) * Dm + colBase + cc],
                      &Kt[r * 64 + cc]);
      uint4 raw = *(const uint4*)&v[(rowBase + key0 + r) * Dm + colBase + cc];
      const half_t* th = (const half_t*)&raw;
      #pragma unroll
      for (int e = 0; e < 8; ++e) Vt[(cc + e) * KT + r] = th[e];
      async_wait_all();
    }
    __syncthreads();

    // scores: two 16x16 tiles (keys key0..+15, key0+16..+31), K-dim = 64
    v8f s0 = zero, s1 = zero;
    #pragma unroll
    for (int ks = 0; ks < 2; ++ks) {
      v16h bk0, bk1;
      const half_t* p0 = &Kt[ln16 * 64 + ks * 32 + (hi << 4)];
      const half_t* p1 = &Kt[(16 + ln16) * 64 + ks * 32 + (hi << 4)];
      #pragma unroll
      for (int vv = 0; vv < 8; ++vv) {
        half2_t h2a = *(const half2_t*)&p0[2 * vv];
        half2_t h2b = *(const half2_t*)&p1[2 * vv];
        bk0[2 * vv] = h2a[0]; bk0[2 * vv + 1] = h2a[1];
        bk1[2 * vv] = h2b[0]; bk1[2 * vv + 1] = h2b[1];
      }
      s0 = __builtin_amdgcn_wmma_f32_16x16x32_f16(false, aq[ks], false, bk0,
                                                  (short)0, s0, false, false);
      s1 = __builtin_amdgcn_wmma_f32_16x16x32_f16(false, aq[ks], false, bk1,
                                                  (short)0, s1, false, false);
    }

    // scale + mask
    const int mk0 = mask[(size_t)b * S + key0 + ln16];
    const int mk1 = mask[(size_t)b * S + key0 + 16 + ln16];
    float f0[8], f1[8];
    #pragma unroll
    for (int r = 0; r < 8; ++r) {
      f0[r] = mk0 ? s0[r] * scale : -1e9f;
      f1[r] = mk1 ? s1[r] * scale : -1e9f;
    }

    // online softmax: row reductions over the 16-lane group
    float alpha_[8];
    #pragma unroll
    for (int r = 0; r < 8; ++r) {
      float t = fmaxf(f0[r], f1[r]);
      #pragma unroll
      for (int off = 8; off; off >>= 1) t = fmaxf(t, __shfl_xor(t, off, 32));
      float mnew = fmaxf(mrun[r], t);
      alpha_[r] = __expf(mrun[r] - mnew);
      mrun[r] = mnew;
      f0[r] = __expf(f0[r] - mnew);
      f1[r] = __expf(f1[r] - mnew);
      float sm = f0[r] + f1[r];
      #pragma unroll
      for (int off = 8; off; off >>= 1) sm += __shfl_xor(sm, off, 32);
      lrun[r] = lrun[r] * alpha_[r] + sm;
    }
    #pragma unroll
    for (int j = 0; j < 4; ++j)
      #pragma unroll
      for (int r = 0; r < 8; ++r) o[j][r] *= alpha_[r];

    // P (C-layout) -> wave-private LDS -> A fragment
    half_t* pb = &Pb[wave * 16 * KT];
    #pragma unroll
    for (int r = 0; r < 8; ++r) {
      int m = r + (hi << 3);
      pb[m * KT + ln16]      = (half_t)f0[r];
      pb[m * KT + 16 + ln16] = (half_t)f1[r];
    }
    v16h ap;
    {
      const half_t* pp = &pb[ln16 * KT];
      #pragma unroll
      for (int vv = 0; vv < 8; ++vv) {
        int kk = (hi << 4) + (vv << 1);
        half2_t h2 = *(const half2_t*)&pp[kk];
        ap[2 * vv] = h2[0]; ap[2 * vv + 1] = h2[1];
      }
    }

    // O += P @ V  (contraction over 32 keys, 4 dim tiles)
    #pragma unroll
    for (int j = 0; j < 4; ++j) {
      v16h bv;
      const half_t* vp2 = &Vt[(j * 16 + ln16) * KT + (hi << 4)];
      #pragma unroll
      for (int vv = 0; vv < 8; ++vv) {
        half2_t h2 = *(const half2_t*)&vp2[2 * vv];
        bv[2 * vv] = h2[0]; bv[2 * vv + 1] = h2[1];
      }
      o[j] = __builtin_amdgcn_wmma_f32_16x16x32_f16(false, ap, false, bv,
                                                    (short)0, o[j], false, false);
    }
  }

  // normalize and store ctx (f16)
  #pragma unroll
  for (int j = 0; j < 4; ++j)
    #pragma unroll
    for (int r = 0; r < 8; ++r) {
      int m = r + (hi << 3);
      float val = o[j][r] / lrun[r];
      ctx[(rowBase + qbase + wave * 16 + m) * Dm + colBase + j * 16 + ln16] =
          (half_t)val;
    }
}

// ---------------------------------------------------------------------------
// host: orchestrate the encoder block
// ---------------------------------------------------------------------------
extern "C" void kernel_launch(void* const* d_in, const int* in_sizes, int n_in,
                              void* d_out, int out_size, void* d_ws, size_t ws_size,
                              hipStream_t stream) {
  constexpr int Bc = 4, Sc = 2048, Dm = 512, Hc = 8, Fm = 2048;
  constexpr int MS = Bc * Sc;  // 8192

  const float* x    = (const float*)d_in[0];
  const int*   msk  = (const int*)d_in[1];
  const float* wq   = (const float*)d_in[2];
  const float* bq   = (const float*)d_in[3];
  const float* wk   = (const float*)d_in[4];
  const float* bk   = (const float*)d_in[5];
  const float* wv   = (const float*)d_in[6];
  const float* bv   = (const float*)d_in[7];
  const float* wo   = (const float*)d_in[8];
  const float* bo   = (const float*)d_in[9];
  const float* w1   = (const float*)d_in[10];
  const float* b1   = (const float*)d_in[11];
  const float* w2   = (const float*)d_in[12];
  const float* b2   = (const float*)d_in[13];
  const float* a1   = (const float*)d_in[14];
  const float* g1   = (const float*)d_in[15];
  const float* a2   = (const float*)d_in[16];
  const float* g2   = (const float*)d_in[17];
  float* out = (float*)d_out;

  char* ws = (char*)d_ws;
  size_t off = 0;
  auto alloc = [&](size_t bytes) -> void* {
    void* p = ws + off;
    off = (off + bytes + 255) & ~(size_t)255;
    return p;
  };

  half_t* h16   = (half_t*)alloc((size_t)MS * Dm * 2);
  half_t* q16   = (half_t*)alloc((size_t)MS * Dm * 2);
  half_t* k16   = (half_t*)alloc((size_t)MS * Dm * 2);
  half_t* v16   = (half_t*)alloc((size_t)MS * Dm * 2);
  half_t* ctx16 = (half_t*)alloc((size_t)MS * Dm * 2);
  half_t* h2b   = (half_t*)alloc((size_t)MS * Dm * 2);
  half_t* ff1   = (half_t*)alloc((size_t)MS * Fm * 2);
  float*  x1    = (float*)alloc((size_t)MS * Dm * 4);
  half_t* wq16  = (half_t*)alloc((size_t)Dm * Dm * 2);
  half_t* wk16  = (half_t*)alloc((size_t)Dm * Dm * 2);
  half_t* wv16  = (half_t*)alloc((size_t)Dm * Dm * 2);
  half_t* wo16  = (half_t*)alloc((size_t)Dm * Dm * 2);
  half_t* w116  = (half_t*)alloc((size_t)Fm * Dm * 2);
  half_t* w216  = (half_t*)alloc((size_t)Dm * Fm * 2);

  // 1) convert weights to f16
  {
    int nDD = Dm * Dm, nFD = Fm * Dm;
    cvt_f32_f16_kernel<<<(nDD + 255) / 256, 256, 0, stream>>>(wq, wq16, nDD);
    cvt_f32_f16_kernel<<<(nDD + 255) / 256, 256, 0, stream>>>(wk, wk16, nDD);
    cvt_f32_f16_kernel<<<(nDD + 255) / 256, 256, 0, stream>>>(wv, wv16, nDD);
    cvt_f32_f16_kernel<<<(nDD + 255) / 256, 256, 0, stream>>>(wo, wo16, nDD);
    cvt_f32_f16_kernel<<<(nFD + 255) / 256, 256, 0, stream>>>(w1, w116, nFD);
    cvt_f32_f16_kernel<<<(nFD + 255) / 256, 256, 0, stream>>>(w2, w216, nFD);
  }

  // 2) LayerNorm 1
  layernorm_f16_kernel<<<MS, 256, 0, stream>>>(x, a1, g1, h16);

  // 3) Q,K,V projections (M=8192, N=512, K=512)
  dim3 gDD(Dm / GBN, MS / GBM);
  gemm_wmma_kernel<<<gDD, 256, 0, stream>>>(h16, wq16, bq, nullptr, nullptr, q16,
                                            MS, Dm, Dm, 0);
  gemm_wmma_kernel<<<gDD, 256, 0, stream>>>(h16, wk16, bk, nullptr, nullptr, k16,
                                            MS, Dm, Dm, 0);
  gemm_wmma_kernel<<<gDD, 256, 0, stream>>>(h16, wv16, bv, nullptr, nullptr, v16,
                                            MS, Dm, Dm, 0);

  // 4) attention
  dim3 gAttn(Sc / 128, Bc * Hc);
  attn_wmma_kernel<<<gAttn, 256, 0, stream>>>(q16, k16, v16, msk, ctx16, Sc, Dm);

  // 5) output projection + residual  -> x1 (fp32)
  gemm_wmma_kernel<<<gDD, 256, 0, stream>>>(ctx16, wo16, bo, x, x1, nullptr,
                                            MS, Dm, Dm, 0);

  // 6) LayerNorm 2
  layernorm_f16_kernel<<<MS, 256, 0, stream>>>(x1, a2, g2, h2b);

  // 7) FFN1 + ReLU (M=8192, N=2048, K=512) -> f16
  dim3 gF1(Fm / GBN, MS / GBM);
  gemm_wmma_kernel<<<gF1, 256, 0, stream>>>(h2b, w116, b1, nullptr, nullptr, ff1,
                                            MS, Fm, Dm, 1);

  // 8) FFN2 + residual (M=8192, N=512, K=2048) -> final fp32 output
  gemm_wmma_kernel<<<gDD, 256, 0, stream>>>(ff1, w216, b2, x1, out, nullptr,
                                            MS, Dm, Fm, 0);
}
